// EdgeGRU_Baseline_NoX_53171695124547
// MI455X (gfx1250) — matrix operand order repaired and checked by reading, not compiled, not run on hardware
//
#include <hip/hip_runtime.h>
#include <hip/hip_bf16.h>

// CDNA5 / gfx1250 wave32 WMMA implementation of the EdgeGRU pipeline.
// All GEMMs run on v_wmma_f32_16x16x32_f16 (f16 inputs, f32 accumulate).
// Edge kernels process 64 edges/block (4 M-tiles) with kt-outer/mt-inner
// loops so each B fragment is reused by 4 WMMAs. h_new is stored as f16
// (classifier consumes it as f16 WMMA input anyway) so the per-edge random
// gathers move half the bytes and need no conversion.
// Workspace layout, ~39 MB:
//   summed[N*128] f32 | cnt[N] f32 | hnewh[N*128] f16 | packed f16 weights

typedef __attribute__((ext_vector_type(16))) _Float16 v16h;
typedef __attribute__((ext_vector_type(8)))  float    v8f;

#define LN_EPS 1e-5f
#define EPB 64   // edges per block (4 WMMA M-tiles)

union V16U { v16h v; float4 q[2]; };

// A-matrix 16x32 f16 fragment (ISA 7.12.2): lane half 0 holds K in
// {0..7, 16..23}, half 1 holds {8..15, 24..31}; within a lane the 16 halves
// are two contiguous 8-half runs -> two aligned 16-byte LDS loads.
__device__ __forceinline__ v16h load_a_frag(const _Float16* rowk, int khalf) {
  V16U u;
  const char* base = (const char*)rowk + khalf * 16;   // + khalf*8 halves
  u.q[0] = *(const float4*)(base);                     // K = khalf*8 .. +7
  u.q[1] = *(const float4*)(base + 32);                // K = khalf*8+16 .. +23
  return u.v;
}

// B fragments are pre-packed so each lane's 16 halves are contiguous.
__device__ __forceinline__ v16h load_b_frag(const _Float16* p) {
  V16U u;
  u.q[0] = ((const float4*)p)[0];
  u.q[1] = ((const float4*)p)[1];
  return u.v;
}

__device__ __forceinline__ v8f wmma32(v16h a, v16h b, v8f c) {
  // (neg_a, A, neg_b, B, c_mod, C, reuse_a, reuse_b)
  return __builtin_amdgcn_wmma_f32_16x16x32_f16(false, a, false, b, (short)0, c,
                                                false, false);
}

__device__ __forceinline__ void atomAddGlobal(float* p, float v) {
  __hip_atomic_fetch_add(p, v, __ATOMIC_RELAXED, __HIP_MEMORY_SCOPE_AGENT);
}
__device__ __forceinline__ void atomAddLds(float* p, float v) {
  __hip_atomic_fetch_add(p, v, __ATOMIC_RELAXED, __HIP_MEMORY_SCOPE_WORKGROUP);
}
__device__ __forceinline__ float sigm(float x) { return 1.0f / (1.0f + __expf(-x)); }

// Parallel per-row LayerNorm stats over a [64 x 128] f32 LDS tile:
// 4 threads per row, each summing 32 elements with float4 loads.
__device__ __forceinline__ void ln_stats(const float* T, float* s_ps, float* s_ps2,
                                         float* s_mu, float* s_rs, int tid) {
  {
    int r = tid >> 2, q = tid & 3;
    const float4* p = (const float4*)&T[r * 128 + q * 32];
    float s = 0.f, s2 = 0.f;
#pragma unroll
    for (int i = 0; i < 8; ++i) {
      float4 v = p[i];
      s  += v.x + v.y + v.z + v.w;
      s2 += v.x * v.x + v.y * v.y + v.z * v.z + v.w * v.w;
    }
    s_ps[tid] = s; s_ps2[tid] = s2;
  }
  __syncthreads();
  if (tid < EPB) {
    float s  = s_ps[tid*4]  + s_ps[tid*4+1]  + s_ps[tid*4+2]  + s_ps[tid*4+3];
    float s2 = s_ps2[tid*4] + s_ps2[tid*4+1] + s_ps2[tid*4+2] + s_ps2[tid*4+3];
    float mu  = s * (1.0f / 128.0f);
    float var = s2 * (1.0f / 128.0f) - mu * mu;
    s_mu[tid] = mu; s_rs[tid] = rsqrtf(var + LN_EPS);
  }
  __syncthreads();
}

// ---------------------------------------------------------------------------
__global__ __launch_bounds__(256)
void k_zero(float* __restrict__ p, unsigned long long n) {
  unsigned long long i = (unsigned long long)blockIdx.x * 256ull + threadIdx.x;
  if (i < n) p[i] = 0.0f;
}

// Pack a [K_total, N_total] (or transposed) fp32 matrix into f16 WMMA
// B-fragments: frag (kt,nt) holds K rows kt*32..+31, N cols nt*16..+15.
// Element (lane,i): kk = kt*32 + (lane>>4)*8 + (i&7) + (i>>3)*16 ; nn = nt*16 + (lane&15).
__global__ __launch_bounds__(256)
void k_pack_b(const float* __restrict__ W, _Float16* __restrict__ dst,
              int KT, int NT, int ldW, int trans) {
  int idx = blockIdx.x * 256 + threadIdx.x;
  int total = KT * NT * 512;
  if (idx >= total) return;
  int i    = idx & 15;
  int lane = (idx >> 4) & 31;
  int frag = idx >> 9;
  int nt = frag % NT, kt = frag / NT;
  int kk = kt * 32 + (lane >> 4) * 8 + (i & 7) + (i >> 3) * 16;
  int nn = nt * 16 + (lane & 15);
  float v = trans ? W[nn * ldW + kk] : W[kk * ldW + nn];
  dst[idx] = (_Float16)v;
}

// ---------------------------------------------------------------------------
// Edge encoder (Lin->LN->ReLU->Lin->LN->ReLU) fused with scatter-add.
// One block = 64 edges; 8 waves, wave w owns output columns [16w, 16w+16)
// for all 4 M-tiles.
__global__ __launch_bounds__(256)
void k_enc_scatter(const float* __restrict__ edge_attr, const int* __restrict__ ei,
                   const float* __restrict__ b1, const float* __restrict__ g1,
                   const float* __restrict__ be1,
                   const float* __restrict__ b2, const float* __restrict__ g2,
                   const float* __restrict__ be2,
                   const _Float16* __restrict__ W1p, const _Float16* __restrict__ W2p,
                   float* __restrict__ summed, float* __restrict__ cnt, int E) {
  __shared__ __align__(16) _Float16 Ah[EPB * 128];
  __shared__ __align__(16) float    T[EPB * 128];
  __shared__ float s_ps[256], s_ps2[256];
  __shared__ float s_mu[EPB], s_rs[EPB];
  __shared__ int   s_src[EPB];

  const int tid = threadIdx.x;
  const int wv = tid >> 5, lane = tid & 31;
  const int m = lane & 15, khalf = lane >> 4, rb = khalf * 8;
  const int e0 = blockIdx.x * EPB;

  // Stage-1 activations: 64x32 edge_attr tile as f16 (ld = 32).
  for (int idx = tid; idx < EPB * 32; idx += 256) {
    int r = idx >> 5, c = idx & 31;
    int e = e0 + r; if (e >= E) e = E - 1;
    Ah[idx] = (_Float16)edge_attr[e * 32 + c];
  }
  if (tid < EPB) {
    int e = e0 + tid; if (e >= E) e = E - 1;
    s_src[tid] = ei[e];
  }
  __syncthreads();

  // GEMM1: [64x32] x [32x128]; B fragment loaded once, used by 4 M-tiles.
  {
    v16h b = load_b_frag(W1p + ((size_t)wv * 32 + lane) * 16);
    int n = wv * 16 + m;
    float bn = b1[n];
#pragma unroll
    for (int mt = 0; mt < 4; ++mt) {
      v16h a = load_a_frag(&Ah[(mt * 16 + m) * 32], khalf);
      v8f c = {};
      c = wmma32(a, b, c);
#pragma unroll
      for (int r = 0; r < 8; ++r) T[(mt * 16 + rb + r) * 128 + n] = c[r] + bn;
    }
  }
  __syncthreads();
  ln_stats(T, s_ps, s_ps2, s_mu, s_rs, tid);
  for (int idx = tid; idx < EPB * 128; idx += 256) {  // LN1 + ReLU -> Ah (ld=128)
    int r = idx >> 7, c = idx & 127;
    float v = (T[idx] - s_mu[r]) * s_rs[r] * g1[c] + be1[c];
    Ah[idx] = (_Float16)fmaxf(v, 0.f);
  }
  __syncthreads();

  // GEMM2: [64x128] x [128x128]; kt-outer, 4 accumulators.
  {
    v8f z = {};
    v8f acc[4] = {z, z, z, z};
#pragma unroll
    for (int kt = 0; kt < 4; ++kt) {
      v16h b = load_b_frag(W2p + ((size_t)(kt * 8 + wv) * 32 + lane) * 16);
#pragma unroll
      for (int mt = 0; mt < 4; ++mt) {
        v16h a = load_a_frag(&Ah[(mt * 16 + m) * 128 + kt * 32], khalf);
        acc[mt] = wmma32(a, b, acc[mt]);
      }
    }
    int n = wv * 16 + m;
    float bn = b2[n];
#pragma unroll
    for (int mt = 0; mt < 4; ++mt)
#pragma unroll
      for (int r = 0; r < 8; ++r) T[(mt * 16 + rb + r) * 128 + n] = acc[mt][r] + bn;
  }
  __syncthreads();
  ln_stats(T, s_ps, s_ps2, s_mu, s_rs, tid);
  // LN2 + ReLU + scatter-add (enc never written to HBM)
  for (int idx = tid; idx < EPB * 128; idx += 256) {
    int r = idx >> 7, c = idx & 127;
    if (e0 + r < E) {
      float v = (T[idx] - s_mu[r]) * s_rs[r] * g2[c] + be2[c];
      atomAddGlobal(&summed[(size_t)s_src[r] * 128 + c], fmaxf(v, 0.f));
    }
  }
  if (tid < EPB && (e0 + tid) < E) atomAddGlobal(&cnt[s_src[tid]], 1.0f);
}

// ---------------------------------------------------------------------------
// GRUCell over 16-node tiles: gx = aggr@Wih.T+bih, gh = h_prev@Whh.T+bhh.
// h_new is emitted as f16 (packed pairs, 4B/lane stores).
__global__ __launch_bounds__(256)
void k_gru(const float* __restrict__ summed, const float* __restrict__ cnt,
           const float* __restrict__ h_prev,
           const _Float16* __restrict__ WihTp, const _Float16* __restrict__ WhhTp,
           const float* __restrict__ bih, const float* __restrict__ bhh,
           _Float16* __restrict__ hnewh, int N) {
  __shared__ __align__(16) _Float16 Aa[16 * 128];
  __shared__ __align__(16) _Float16 Ahp[16 * 128];
  __shared__ float Gx[16 * 384];
  __shared__ float Gh[16 * 384];
  __shared__ float Hp[16 * 128];

  const int tid = threadIdx.x;
  const int wv = tid >> 5, lane = tid & 31;
  const int m = lane & 15, khalf = lane >> 4;
  const int n0 = blockIdx.x * 16;

  for (int idx = tid; idx < 2048; idx += 256) {
    int r = idx >> 7, c = idx & 127;
    int node = n0 + r; if (node >= N) node = N - 1;
    float ct = fmaxf(cnt[node], 1.0f);
    float ag = summed[(size_t)node * 128 + c] / ct;
    Aa[idx] = (_Float16)ag;
    float hp = h_prev[(size_t)node * 128 + c];
    Hp[idx] = hp;
    Ahp[idx] = (_Float16)hp;
  }
  __syncthreads();

  // 48 output tiles (24 gx + 24 gh), 6 per wave, 4 K-steps each.
  for (int t = wv; t < 48; t += 8) {
    bool isX = t < 24;
    int nt = isX ? t : t - 24;
    const _Float16* Asrc = isX ? Aa : Ahp;
    const _Float16* Wp   = isX ? WihTp : WhhTp;
    const float*    bias = isX ? bih : bhh;
    float*          G    = isX ? Gx : Gh;
    v8f c = {};
#pragma unroll
    for (int kt = 0; kt < 4; ++kt) {
      v16h a = load_a_frag(&Asrc[m * 128 + kt * 32], khalf);
      v16h b = load_b_frag(Wp + ((size_t)(kt * 24 + nt) * 32 + lane) * 16);
      c = wmma32(a, b, c);
    }
    int n = nt * 16 + m;
    float bn = bias[n];
    int rb = khalf * 8;
#pragma unroll
    for (int r = 0; r < 8; ++r) G[(rb + r) * 384 + n] = c[r] + bn;
  }
  __syncthreads();

  // Gates (order r, z, n); two adjacent channels per thread -> packed b32 store.
  for (int idx = tid; idx < 1024; idx += 256) {
    int r = idx >> 6, p = idx & 63;
    int cc = p * 2;
    int node = n0 + r;
    if (node < N) {
      union { _Float16 h[2]; unsigned u; } pk;
#pragma unroll
      for (int j = 0; j < 2; ++j) {
        int c = cc + j;
        float xr = Gx[r * 384 + c], xz = Gx[r * 384 + 128 + c], xn = Gx[r * 384 + 256 + c];
        float hr = Gh[r * 384 + c], hz = Gh[r * 384 + 128 + c], hn = Gh[r * 384 + 256 + c];
        float rg = sigm(xr + hr);
        float zg = sigm(xz + hz);
        float ng = tanhf(xn + rg * hn);
        pk.h[j] = (_Float16)((1.0f - zg) * ng + zg * Hp[r * 128 + c]);
      }
      *(unsigned*)&hnewh[(size_t)node * 128 + cc] = pk.u;
    }
  }
}

// ---------------------------------------------------------------------------
// Classifier: relu([h_src | h_dst | edge_attr] @ Wc1 + bc1) @ Wc2 + bc2.
// One block = 64 edges; kt-outer (K=288 -> 9 steps), 4 accumulators.
// h_new gathers are pure f16 16-byte copies (no conversion).
__global__ __launch_bounds__(256)
void k_classifier(const int* __restrict__ ei, const float* __restrict__ edge_attr,
                  const _Float16* __restrict__ hnewh,
                  const _Float16* __restrict__ Wc1p, const float* __restrict__ bc1,
                  const float* __restrict__ Wc2, const float* __restrict__ bc2,
                  float* __restrict__ out, int E) {
  __shared__ __align__(16) _Float16 Ah[EPB * 288];
  __shared__ int   s_src[EPB], s_dst[EPB];
  __shared__ float s_out[EPB];

  const int tid = threadIdx.x;
  const int wv = tid >> 5, lane = tid & 31;
  const int m = lane & 15, khalf = lane >> 4, rb = khalf * 8;
  const int e0 = blockIdx.x * EPB;

  // Overlap: prefetch this block's 8KB edge_attr tile (read last in the
  // gather phase, behind the ei -> h_new dependency chain).
  if (e0 + EPB <= E) {
    const char* p = (const char*)edge_attr + (size_t)e0 * 128 + (size_t)tid * 32;
    __builtin_prefetch(p, 0, 3);
  }

  if (tid < EPB) {
    int e = e0 + tid; if (e >= E) e = E - 1;
    s_src[tid] = ei[e];
    s_dst[tid] = ei[E + e];
    s_out[tid] = 0.f;
  }
  __syncthreads();
  // Gather the 64x288 activation tile. h_src / h_dst rows: 16B f16 copies.
  for (int idx = tid; idx < EPB * 16; idx += 256) {
    int r = idx >> 4, q = idx & 15;
    *(float4*)&Ah[r * 288 + q * 8] =
        *(const float4*)&hnewh[(size_t)s_src[r] * 128 + q * 8];
  }
  for (int idx = tid; idx < EPB * 16; idx += 256) {
    int r = idx >> 4, q = idx & 15;
    *(float4*)&Ah[r * 288 + 128 + q * 8] =
        *(const float4*)&hnewh[(size_t)s_dst[r] * 128 + q * 8];
  }
  for (int idx = tid; idx < EPB * 32; idx += 256) {
    int r = idx >> 5, c = idx & 31;
    int e = e0 + r; if (e >= E) e = E - 1;
    Ah[r * 288 + 256 + c] = (_Float16)edge_attr[e * 32 + c];
  }
  __syncthreads();

  v8f z = {};
  v8f acc[4] = {z, z, z, z};
#pragma unroll
  for (int kt = 0; kt < 9; ++kt) {  // K = 288
    v16h b = load_b_frag(Wc1p + ((size_t)(kt * 8 + wv) * 32 + lane) * 16);
#pragma unroll
    for (int mt = 0; mt < 4; ++mt) {
      v16h a = load_a_frag(&Ah[(mt * 16 + m) * 288 + kt * 32], khalf);
      acc[mt] = wmma32(a, b, acc[mt]);
    }
  }
  int n = wv * 16 + m;
  float bn = bc1[n], wn = Wc2[n];
#pragma unroll
  for (int mt = 0; mt < 4; ++mt)
#pragma unroll
    for (int r = 0; r < 8; ++r) {
      float v = fmaxf(acc[mt][r] + bn, 0.f) * wn;
      atomAddLds(&s_out[mt * 16 + rb + r], v);
    }
  __syncthreads();
  if (tid < EPB && (e0 + tid) < E) out[e0 + tid] = s_out[tid] + bc2[0];
}

// ---------------------------------------------------------------------------
extern "C" void kernel_launch(void* const* d_in, const int* in_sizes, int n_in,
                              void* d_out, int out_size, void* d_ws, size_t ws_size,
                              hipStream_t stream) {
  (void)n_in; (void)out_size; (void)ws_size;
  const int*   ei        = (const int*)  d_in[1];
  const float* edge_attr = (const float*)d_in[2];
  const float* h_prev    = (const float*)d_in[3];
  const float* W1  = (const float*)d_in[4];
  const float* b1  = (const float*)d_in[5];
  const float* g1  = (const float*)d_in[6];
  const float* be1 = (const float*)d_in[7];
  const float* W2  = (const float*)d_in[8];
  const float* b2  = (const float*)d_in[9];
  const float* g2  = (const float*)d_in[10];
  const float* be2 = (const float*)d_in[11];
  const float* Wih = (const float*)d_in[12];
  const float* bih = (const float*)d_in[13];
  const float* Whh = (const float*)d_in[14];
  const float* bhh = (const float*)d_in[15];
  const float* Wc1 = (const float*)d_in[16];
  const float* bc1 = (const float*)d_in[17];
  const float* Wc2 = (const float*)d_in[18];
  const float* bc2 = (const float*)d_in[19];
  float* out = (float*)d_out;

  const int E = in_sizes[2] / 32;    // 500000
  const int N = in_sizes[3] / 128;   // 50000

  // Workspace carve-up (~39 MB).
  float*    summed = (float*)d_ws;                        // N*128 f32
  float*    cnt    = summed + (size_t)N * 128;            // N f32
  _Float16* hnewh  = (_Float16*)(cnt + N);                // N*128 f16
  _Float16* W1p    = hnewh + (size_t)N * 128;
  _Float16* W2p    = W1p   + 1 * 8 * 512;
  _Float16* WihTp  = W2p   + 4 * 8 * 512;
  _Float16* WhhTp  = WihTp + 4 * 24 * 512;
  _Float16* Wc1p   = WhhTp + 4 * 24 * 512;

  // Zero scatter accumulators (summed and cnt are contiguous).
  {
    unsigned long long nz = (unsigned long long)N * 128ull + (unsigned long long)N;
    k_zero<<<(unsigned)((nz + 255) / 256), 256, 0, stream>>>(summed, nz);
  }
  // Repack weights into WMMA f16 B-fragments (tiny vs. edge work).
  k_pack_b<<<(1 * 8  * 512 + 255) / 256, 256, 0, stream>>>(W1,  W1p,   1, 8,  128, 0);
  k_pack_b<<<(4 * 8  * 512 + 255) / 256, 256, 0, stream>>>(W2,  W2p,   4, 8,  128, 0);
  k_pack_b<<<(4 * 24 * 512 + 255) / 256, 256, 0, stream>>>(Wih, WihTp, 4, 24, 128, 1);
  k_pack_b<<<(4 * 24 * 512 + 255) / 256, 256, 0, stream>>>(Whh, WhhTp, 4, 24, 128, 1);
  k_pack_b<<<(9 * 8  * 512 + 255) / 256, 256, 0, stream>>>(Wc1, Wc1p,  9, 8,  128, 0);

  const int EB = (E + EPB - 1) / EPB, NB = (N + 15) / 16;
  k_enc_scatter<<<EB, 256, 0, stream>>>(edge_attr, ei, b1, g1, be1, b2, g2, be2,
                                        W1p, W2p, summed, cnt, E);
  k_gru<<<NB, 256, 0, stream>>>(summed, cnt, h_prev, WihTp, WhhTp, bih, bhh, hnewh, N);
  k_classifier<<<EB, 256, 0, stream>>>(ei, edge_attr, hnewh, Wc1p, bc1, Wc2, bc2, out, E);
}
